// SkipGenerator_67877663146004
// MI455X (gfx1250) — matrix operand order repaired and checked by reading, not compiled, not run on hardware
//
#include <hip/hip_runtime.h>

typedef __attribute__((ext_vector_type(16))) __bf16 v16bf;
typedef __attribute__((ext_vector_type(8)))  __bf16 v8bf;
typedef __attribute__((ext_vector_type(8)))  float  v8f;

// ---------------------------------------------------------------------------
// Zero a small scratch block (used as the padding source for A-fragment loads).
// ---------------------------------------------------------------------------
__global__ void zeroblk_kernel(__bf16* __restrict__ z) {
  z[threadIdx.x] = (__bf16)0.f;
}

// ---------------------------------------------------------------------------
// Style affine: s[b][ci] = latents[b,row,:] @ sw[:,ci] + sb[ci]
// ---------------------------------------------------------------------------
__global__ void style_kernel(const float* __restrict__ latents, int row,
                             const float* __restrict__ sw,
                             const float* __restrict__ sb,
                             float* __restrict__ sout, int Cin) {
  int b = blockIdx.x;
  const float* lat = latents + ((size_t)b * 10 + row) * 512;
  for (int ci = threadIdx.x; ci < Cin; ci += blockDim.x) {
    float acc = sb[ci];
    for (int l = 0; l < 512; ++l) acc += lat[l] * sw[(size_t)l * Cin + ci];
    sout[(size_t)b * Cin + ci] = acc;
  }
}

// ---------------------------------------------------------------------------
// Demodulation: d[b][co] = rsqrt(sum_{tap,ci} (w * s)^2 + 1e-8)
// ---------------------------------------------------------------------------
__global__ void demod_kernel(const float* __restrict__ w,
                             const float* __restrict__ s,
                             float* __restrict__ dout,
                             int Cin, int Cout, int B) {
  int idx = blockIdx.x * blockDim.x + threadIdx.x;
  if (idx >= B * Cout) return;
  int b = idx / Cout, co = idx - b * Cout;
  float sum = 0.f;
  for (int t = 0; t < 9; ++t)
    for (int ci = 0; ci < Cin; ++ci) {
      float v = w[(size_t)(t * Cin + ci) * Cout + co] * s[(size_t)b * Cin + ci];
      sum += v * v;
    }
  dout[idx] = rsqrtf(sum + 1e-8f);
}

// ---------------------------------------------------------------------------
// FIR 2x upsample (upfirdn2d, k = outer([1,3,3,1])/16): (B,H,W,C)->(B,2H,2W,C)
// ---------------------------------------------------------------------------
__global__ void blur_up_kernel(const float* __restrict__ x,
                               float* __restrict__ out,
                               int H, int W, int C, int B, int accumulate) {
  int OH = 2 * H, OW = 2 * W;
  int total = B * OH * OW * C;
  int idx = blockIdx.x * blockDim.x + threadIdx.x;
  if (idx >= total) return;
  int c = idx % C;
  int t = idx / C;
  int ox = t % OW; t /= OW;
  int oy = t % OH; int b = t / OH;
  const float k1[4] = {1.f, 3.f, 3.f, 1.f};
  float acc = 0.f;
  for (int ky = 0; ky < 4; ++ky) {
    int py = oy + ky - 2;
    if (py < 0 || (py & 1)) continue;
    int iy = py >> 1; if (iy >= H) continue;
    for (int kx = 0; kx < 4; ++kx) {
      int px = ox + kx - 2;
      if (px < 0 || (px & 1)) continue;
      int ix = px >> 1; if (ix >= W) continue;
      acc += k1[ky] * k1[kx] * x[(((size_t)b * H + iy) * W + ix) * C + c];
    }
  }
  acc *= (1.f / 16.f);
  if (accumulate) out[idx] += acc; else out[idx] = acc;
}

// ---------------------------------------------------------------------------
// 1x1 to-RGB (no demod): out[b,p,0..2] = leaky(sum_ci x*s*w + bias)
// ---------------------------------------------------------------------------
__global__ void rgb_kernel(const float* __restrict__ x,
                           const float* __restrict__ s,
                           const float* __restrict__ w,
                           const float* __restrict__ bias,
                           float* __restrict__ out, int HW, int Cin, int B) {
  int idx = blockIdx.x * blockDim.x + threadIdx.x;
  if (idx >= B * HW) return;
  int b = idx / HW;
  const float* xp = x + (size_t)idx * Cin;
  const float* sp = s + (size_t)b * Cin;
  float a0 = 0.f, a1 = 0.f, a2 = 0.f;
  for (int ci = 0; ci < Cin; ++ci) {
    float v = xp[ci] * sp[ci];
    a0 += v * w[ci * 3 + 0];
    a1 += v * w[ci * 3 + 1];
    a2 += v * w[ci * 3 + 2];
  }
  a0 += bias[0]; a1 += bias[1]; a2 += bias[2];
  a0 = a0 > 0.f ? a0 : 0.01f * a0;
  a1 = a1 > 0.f ? a1 : 0.01f * a1;
  a2 = a2 > 0.f ? a2 : 0.01f * a2;
  out[(size_t)idx * 3 + 0] = a0;
  out[(size_t)idx * 3 + 1] = a1;
  out[(size_t)idx * 3 + 2] = a2;
}

// ---------------------------------------------------------------------------
// Pre-modulate + convert: xbf[b,p,ci] = bf16(x[b,p,ci] * s[b,ci])
// (xBS = 0 broadcasts the const input across batch)
// ---------------------------------------------------------------------------
__global__ void premod_kernel(const float* __restrict__ x, long xBS,
                              const float* __restrict__ s,
                              __bf16* __restrict__ o, int HW, int Cin, int B) {
  int idx = blockIdx.x * blockDim.x + threadIdx.x;
  if (idx >= B * HW * Cin) return;
  int ci = idx % Cin;
  int t  = idx / Cin;
  int p  = t % HW;
  int b  = t / HW;
  o[idx] = (__bf16)(x[(size_t)b * xBS + (size_t)p * Cin + ci] *
                    s[(size_t)b * Cin + ci]);
}

// ---------------------------------------------------------------------------
// Swizzle weights (3,3,Cin,Cout) f32 -> bf16 in exact WMMA B-fragment layout:
//   o[((kb*NT16 + nb)*32 + lane)*16 + j] = w[kb*32 + (lane>=16)*8 + (j&7) + (j>>3)*16, nb*16 + (lane&15)]
// so each lane's 16-element B fragment is one contiguous 32B load.
// ---------------------------------------------------------------------------
__global__ void wswizzle_kernel(const float* __restrict__ w,
                                __bf16* __restrict__ o, int Cin, int Cout) {
  int NT16 = Cout >> 4;
  int total = ((9 * Cin) >> 5) * NT16 * 512;
  int idx = blockIdx.x * blockDim.x + threadIdx.x;
  if (idx >= total) return;
  int j    = idx & 15;
  int lane = (idx >> 4) & 31;
  int t    = idx >> 9;            // kb*NT16 + nb
  int nb = t % NT16, kb = t / NT16;
  int k  = kb * 32 + ((lane >= 16) ? 8 : 0) + (j & 7) + ((j >> 3) << 4);
  int co = nb * 16 + (lane & 15);
  o[idx] = (__bf16)w[(size_t)k * Cout + co];
}

// ---------------------------------------------------------------------------
// Implicit-GEMM modulated 3x3 SAME conv, WMMA bf16 / f32-accum, LDS-free.
// Block = 128 threads = 4 waves; block tile M=16 x N=128; each wave owns two
// 16x16 N-tiles sharing one A fragment.
// Loop structure: outer over the 9 taps (padding validity + row base are
// tap-invariant, hoisted), inner over Cin/32 K-steps with pure pointer
// increments. Out-of-bounds A rows read from a zeroed scratch block selected
// by v_cndmask on the pointer -- no control flow in the inner loop.
// ---------------------------------------------------------------------------
__global__ __launch_bounds__(128) void conv3x3_wmma(
    const __bf16* __restrict__ xbf, long xBS, int H, int wShift, int cinShift,
    const __bf16* __restrict__ wsw, const __bf16* __restrict__ zblk,
    const float* __restrict__ bias, int Cout,
    const float* __restrict__ d, const float* __restrict__ noise,
    const float* __restrict__ nsptr, int applyAct,
    float* __restrict__ out, int Mtiles, int NtBlocks) {
  int tid = threadIdx.x, lane = tid & 31, wv = tid >> 5;
  int bi  = blockIdx.x;
  int b   = bi / (Mtiles * NtBlocks);
  int rem = bi - b * (Mtiles * NtBlocks);
  int mt  = rem / NtBlocks;
  int ntb = rem - mt * NtBlocks;

  int Cin  = 1 << cinShift;
  int NT16 = Cout >> 4;
  int HW   = H * H;

  int hi  = (lane >= 16);
  int mL  = lane & 15;
  int hi8 = hi << 3;

  // Per-lane spatial row of the A fragment (fixed for the whole kernel).
  int m0 = mt * 16;
  int p  = m0 + mL;
  int oy = p >> wShift;
  int ox = p & (H - 1);

  const __bf16* rowbase = xbf + (size_t)b * xBS + hi8;

  // Two B-fragment streams for this wave (advance linearly across all taps).
  size_t wstep = (size_t)NT16 * 512;
  const __bf16* wq0 = wsw + (size_t)(ntb * 8 + (wv << 1)) * 512 + lane * 16;
  const __bf16* wq1 = wq0 + 512;

  v8f c0 = {}, c1 = {};
  int inner = Cin >> 5;   // K-steps per tap (4, 8 or 16)

  for (int tap = 0; tap < 9; ++tap) {
    int ky = (tap * 11) >> 5;          // tap / 3 for tap in [0,8]
    int kx = tap - ky * 3;
    int iy = oy + ky - 1;
    int ix = ox + kx - 1;
    bool valid = ((unsigned)iy < (unsigned)H) & ((unsigned)ix < (unsigned)H);
    const __bf16* ap = rowbase + (long)(iy * H + ix) * Cin;
#pragma unroll 4
    for (int t = 0; t < inner; ++t) {
      const __bf16* sel = valid ? ap : zblk;
      v8bf a0 = *(const v8bf*)sel;
      v8bf a1 = *(const v8bf*)(sel + 16);
      ap += 32;
      v16bf af = __builtin_shufflevector(a0, a1, 0, 1, 2, 3, 4, 5, 6, 7,
                                         8, 9, 10, 11, 12, 13, 14, 15);
      v16bf bf0 = *(const v16bf*)wq0;
      v16bf bf1 = *(const v16bf*)wq1;
      __builtin_prefetch(wq0 + wstep, 0, 1);
      wq0 += wstep;
      wq1 += wstep;
      c0 = __builtin_amdgcn_wmma_f32_16x16x32_bf16(false, af, false, bf0,
                                                   (short)0, c0, false, false);
      c1 = __builtin_amdgcn_wmma_f32_16x16x32_bf16(false, af, false, bf1,
                                                   (short)0, c1, false, false);
    }
  }

  // Epilogue. C/D layout: VGPR r -> M = r (+8 for hi lanes); N = lane&15.
  int co0 = ntb * 128 + (wv << 5) + mL;
  int co1 = co0 + 16;
  float d0 = d ? d[(size_t)b * Cout + co0] : 1.f;
  float d1 = d ? d[(size_t)b * Cout + co1] : 1.f;
  float b0 = bias[co0], b1 = bias[co1];
  float ns = nsptr ? nsptr[0] : 0.f;
  int hir = hi ? 8 : 0;
#pragma unroll
  for (int r = 0; r < 8; ++r) {
    int pr = m0 + r + hir;
    float nz = noise ? ns * noise[(size_t)b * HW + pr] : 0.f;
    float v0 = c0[r] * d0 + b0 + nz;
    float v1 = c1[r] * d1 + b1 + nz;
    if (applyAct) {
      v0 = v0 > 0.f ? v0 : 0.01f * v0;
      v1 = v1 > 0.f ? v1 : 0.01f * v1;
    }
    size_t ob = ((size_t)b * HW + pr) * Cout;
    out[ob + co0] = v0;
    out[ob + co1] = v1;
  }
}

// ---------------------------------------------------------------------------
// Host orchestration
// ---------------------------------------------------------------------------
static inline int ilog2i(int x) { int s = 0; while ((1 << s) < x) ++s; return s; }

extern "C" void kernel_launch(void* const* d_in, const int* in_sizes, int n_in,
                              void* d_out, int out_size, void* d_ws, size_t ws_size,
                              hipStream_t stream) {
  (void)out_size; (void)ws_size;
  const float* in[60];
  for (int i = 0; i < n_in && i < 60; ++i) in[i] = (const float*)d_in[i];
  int B = in_sizes[0] / (10 * 512);

  // Workspace layout (bytes); zero block first for 16B alignment.
  char* base = (char*)d_ws;
  size_t off = 0;
  __bf16* zblk = (__bf16*)(base + off); off += 64 * 2;
  float* S = (float*)(base + off); off += (size_t)13 * B * 512 * 4;
  float* D = (float*)(base + off); off += (size_t)9 * B * 512 * 4;
  size_t fcap = (size_t)B * 64 * 64 * 512;
  float* yA = (float*)(base + off); off += fcap * 4;
  float* yB = (float*)(base + off); off += fcap * 4;
  float* yT = (float*)(base + off); off += fcap * 4;
  float* rgbA = (float*)(base + off); off += (size_t)B * 64 * 64 * 3 * 4;
  float* rgbB = (float*)(base + off); off += (size_t)B * 64 * 64 * 3 * 4;
  __bf16* xbf = (__bf16*)(base + off); off += fcap * 2;                 // premod bf16
  __bf16* wbf = (__bf16*)(base + off); off += (size_t)4608 * 512 * 2;   // swizzled wts

  const int cs[5] = {64, 128, 256, 512, 512};

  zeroblk_kernel<<<1, 64, 0, stream>>>(zblk);

  // --- styles ---
  style_kernel<<<B, 256, 0, stream>>>(in[0], 0, in[4],  in[5],  S,                       512);
  style_kernel<<<B, 256, 0, stream>>>(in[0], 5, in[10], in[11], S + (size_t)1 * B * 512, 512);
  {
    int cprev = 512;
    for (int i = 1; i <= 4; ++i) {
      int c = cs[i];
      int bx = 12 + (i - 1) * 12;
      style_kernel<<<B, 256, 0, stream>>>(in[0], 4 + i, in[bx + 2],  in[bx + 3],
                                          S + (size_t)(2 + 3 * (i - 1)) * B * 512, cprev);
      style_kernel<<<B, 256, 0, stream>>>(in[0], i,     in[bx + 6],  in[bx + 7],
                                          S + (size_t)(3 + 3 * (i - 1)) * B * 512, c);
      style_kernel<<<B, 256, 0, stream>>>(in[0], 5 + i, in[bx + 10], in[bx + 11],
                                          S + (size_t)(4 + 3 * (i - 1)) * B * 512, c);
      cprev = c;
    }
  }

  // --- demod factors ---
  {
    int tot = B * 512;
    demod_kernel<<<(tot + 255) / 256, 256, 0, stream>>>(in[2], S, D, 512, 512, B);
    int cprev = 512;
    for (int i = 1; i <= 4; ++i) {
      int c = cs[i];
      int bx = 12 + (i - 1) * 12;
      int tu = B * c;
      demod_kernel<<<(tu + 255) / 256, 256, 0, stream>>>(
          in[bx],     S + (size_t)(2 + 3 * (i - 1)) * B * 512,
          D + (size_t)(1 + 2 * (i - 1)) * B * 512, cprev, c, B);
      demod_kernel<<<(tu + 255) / 256, 256, 0, stream>>>(
          in[bx + 4], S + (size_t)(3 + 3 * (i - 1)) * B * 512,
          D + (size_t)(2 + 2 * (i - 1)) * B * 512, c, c, B);
      cprev = c;
    }
  }

  // --- conv0: 4x4, const input, demod + noise + leaky ---
  {
    int H = 4, Cin = 512, Cout = 512, HW = 16;
    int tp = B * HW * Cin;
    premod_kernel<<<(tp + 255) / 256, 256, 0, stream>>>(in[1], 0L, S, xbf, HW, Cin, B);
    int tw = 9 * Cin * Cout;
    wswizzle_kernel<<<(tw + 255) / 256, 256, 0, stream>>>(in[2], wbf, Cin, Cout);
    int Mt = HW / 16, Nt = Cout / 128;
    conv3x3_wmma<<<B * Mt * Nt, 128, 0, stream>>>(
        xbf, (long)HW * Cin, H, ilog2i(H), ilog2i(Cin), wbf, zblk, in[3], Cout,
        D, in[7], in[6], 1, yA, Mt, Nt);
  }
  // --- rgb0 ---
  rgb_kernel<<<(B * 16 + 255) / 256, 256, 0, stream>>>(
      yA, S + (size_t)1 * B * 512, in[8], in[9], rgbA, 16, 512, B);

  // --- levels 1..4 ---
  float* curRgb = rgbA;
  int Hin = 4, cprev = 512;
  for (int i = 1; i <= 4; ++i) {
    int c = cs[i];
    int Ho = Hin * 2, HW = Ho * Ho;
    int bx = 12 + (i - 1) * 12;
    const float* Su = S + (size_t)(2 + 3 * (i - 1)) * B * 512;
    const float* Sc = S + (size_t)(3 + 3 * (i - 1)) * B * 512;
    const float* Sr = S + (size_t)(4 + 3 * (i - 1)) * B * 512;
    const float* Du = D + (size_t)(1 + 2 * (i - 1)) * B * 512;
    const float* Dc = D + (size_t)(2 + 2 * (i - 1)) * B * 512;

    // FIR 2x upsample of features
    {
      int tot = B * HW * cprev;
      blur_up_kernel<<<(tot + 255) / 256, 256, 0, stream>>>(yA, yT, Hin, Hin, cprev, B, 0);
    }
    // UpsampleConv 3x3 (demod)
    {
      int tp = B * HW * cprev;
      premod_kernel<<<(tp + 255) / 256, 256, 0, stream>>>(yT, (long)HW * cprev, Su, xbf, HW, cprev, B);
      int tw = 9 * cprev * c;
      wswizzle_kernel<<<(tw + 255) / 256, 256, 0, stream>>>(in[bx], wbf, cprev, c);
      int Mt = HW / 16, Nt = c / 128;
      conv3x3_wmma<<<B * Mt * Nt, 128, 0, stream>>>(
          xbf, (long)HW * cprev, Ho, ilog2i(Ho), ilog2i(cprev), wbf, zblk, in[bx + 1], c,
          Du, nullptr, nullptr, 0, yB, Mt, Nt);
    }
    // Plain modulated 3x3 (demod)
    {
      int tp = B * HW * c;
      premod_kernel<<<(tp + 255) / 256, 256, 0, stream>>>(yB, (long)HW * c, Sc, xbf, HW, c, B);
      int tw = 9 * c * c;
      wswizzle_kernel<<<(tw + 255) / 256, 256, 0, stream>>>(in[bx + 4], wbf, c, c);
      int Mt = HW / 16, Nt = c / 128;
      conv3x3_wmma<<<B * Mt * Nt, 128, 0, stream>>>(
          xbf, (long)HW * c, Ho, ilog2i(Ho), ilog2i(c), wbf, zblk, in[bx + 5], c,
          Dc, nullptr, nullptr, 0, yA, Mt, Nt);
    }
    // to_rgb + skip accumulation of blurred previous rgb
    float* tgt = (i == 4) ? (float*)d_out : ((curRgb == rgbA) ? rgbB : rgbA);
    rgb_kernel<<<(B * HW + 255) / 256, 256, 0, stream>>>(
        yA, Sr, in[bx + 8], in[bx + 9], tgt, HW, c, B);
    {
      int tot = B * HW * 3;
      blur_up_kernel<<<(tot + 255) / 256, 256, 0, stream>>>(curRgb, tgt, Hin, Hin, 3, B, 1);
    }
    curRgb = tgt;
    Hin = Ho;
    cprev = c;
  }
}
